// Net_62929860821387
// MI455X (gfx1250) — matrix operand order repaired and checked by reading, not compiled, not run on hardware
//
#include <hip/hip_runtime.h>
#include <hip/hip_bf16.h>

#define NN 50000
#define NE 800000
#define FI 128
#define HD 256
#define CD 32
#define EPSF 1e-12f

typedef __attribute__((ext_vector_type(16))) __bf16 bf16x16;
typedef __attribute__((ext_vector_type(8)))  __bf16 bf16x8;
typedef __attribute__((ext_vector_type(8)))  float  f32x8;

// ---------------- WMMA helpers (CDNA5 16x16x32 bf16, f32 accum) ----------------

__device__ __forceinline__ f32x8 wmma_bf(bf16x16 a, bf16x16 b, f32x8 c) {
  return __builtin_amdgcn_wmma_f32_16x16x32_bf16(false, a, false, b, (short)0, c,
                                                 false, false);
}

// A fragment: 16x32 bf16 tile, row-major source (LDS or global), lda in elems.
// Lane L holds row (L&15); lanes 0-15: K=0..7,16..23; lanes 16-31: K=8..15,24..31.
__device__ __forceinline__ bf16x16 load_a_frag(const __bf16* base, int lda, int lane) {
  const int row = lane & 15, half = lane >> 4;
  const __bf16* p = base + (size_t)row * lda + half * 8;
  bf16x8 lo = *(const bf16x8*)(p);
  bf16x8 hi = *(const bf16x8*)(p + 16);
  return __builtin_shufflevector(lo, hi, 0,1,2,3,4,5,6,7,8,9,10,11,12,13,14,15);
}

// B fragment: 32x16 bf16 tile from an N-major (pre-transposed) weight Wt[N][K].
// Lane L holds col (L&15); lanes 0-15: K=0..7,16..23; lanes 16-31: K=8..15,24..31.
// -> two 16B loads per lane.
__device__ __forceinline__ bf16x16 load_b_frag(const __bf16* wt, int ldk,
                                               int n0, int k0, int lane) {
  const int col = lane & 15, half = lane >> 4;
  const __bf16* p = wt + (size_t)(n0 + col) * ldk + k0 + half * 8;
  bf16x8 lo = *(const bf16x8*)(p);
  bf16x8 hi = *(const bf16x8*)(p + 16);
  return __builtin_shufflevector(lo, hi, 0,1,2,3,4,5,6,7,8,9,10,11,12,13,14,15);
}

// ---------------- Utility kernels ----------------

__global__ void zero_f32(float* p, long n) {
  long i = (long)blockIdx.x * blockDim.x + threadIdx.x;
  long stride = (long)gridDim.x * blockDim.x;
  for (; i < n; i += stride) p[i] = 0.0f;
}

// W[K][N] f32 -> Wt[N][K] bf16 (N-major so B-fragments are contiguous loads)
__global__ void w_to_bf16_t(const float* __restrict__ w, __bf16* __restrict__ wt,
                            int K, int N) {
  int i = blockIdx.x * blockDim.x + threadIdx.x;
  if (i < K * N) {
    int k = i / N, n = i - k * N;
    wt[(size_t)n * K + k] = (__bf16)w[i];
  }
}

__global__ void count_edges(const int* __restrict__ dst, float* __restrict__ cnt) {
  int e = blockIdx.x * blockDim.x + threadIdx.x;
  if (e < NE) atomicAdd(&cnt[dst[e]], 1.0f);
}

// ---------------- Layer 1 fused edge kernel ----------------
// 32 edges/block, 256 threads (8 waves). N = K = 128.
__global__ __launch_bounds__(256) void l1_edge(
    const float* __restrict__ node_feat,      // [NN,128]
    const float* __restrict__ edge_feat,      // [NE,2]
    const int* __restrict__ src, const int* __restrict__ dst,
    const float* __restrict__ We1, const float* __restrict__ be1,   // [2,128],[128]
    const __bf16* __restrict__ Wr1a_t, const float* __restrict__ br1a,
    const __bf16* __restrict__ Wr1b_t, const float* __restrict__ br1b,
    float* __restrict__ s1,                   // [NN,128] accum
    __bf16* __restrict__ e_h)                 // [NE,128] relu(edge transform)
{
  __shared__ __bf16 mtile[32][FI];   // 8 KB
  __shared__ __bf16 rtile[32][FI];   // 8 KB
  __shared__ int    sdst[32];

  const int tid   = threadIdx.x;
  const int ebase = blockIdx.x * 32;

  // Phase A: per-edge cosine + message m = cos * w * hs  (8 threads/edge)
  {
    const int el = tid >> 3, part = tid & 7;     // 16 cols each
    const int e  = ebase + el;
    const int s  = src[e], d = dst[e];
    if (part == 0) sdst[el] = d;
    const float e0 = edge_feat[2 * e], e1 = edge_feat[2 * e + 1];
    const float* hs = node_feat + (size_t)s * FI + part * 16;
    const float* hd = node_feat + (size_t)d * FI + part * 16;
    float hv[16];
    float ss = 0.f, dd = 0.f, sd = 0.f;
#pragma unroll
    for (int j = 0; j < 16; j += 4) {
      float4 a = *(const float4*)(hs + j);
      float4 b = *(const float4*)(hd + j);
      hv[j] = a.x; hv[j+1] = a.y; hv[j+2] = a.z; hv[j+3] = a.w;
      ss += a.x*a.x + a.y*a.y + a.z*a.z + a.w*a.w;
      dd += b.x*b.x + b.y*b.y + b.z*b.z + b.w*b.w;
      sd += a.x*b.x + a.y*b.y + a.z*b.z + a.w*b.w;
    }
#pragma unroll
    for (int m = 1; m < 8; m <<= 1) {
      ss += __shfl_xor(ss, m, 32);
      dd += __shfl_xor(dd, m, 32);
      sd += __shfl_xor(sd, m, 32);
    }
    const float cosv = sd / (fmaxf(sqrtf(ss), EPSF) * fmaxf(sqrtf(dd), EPSF));
#pragma unroll
    for (int j = 0; j < 16; ++j) {
      const int c = part * 16 + j;
      const float w = e0 * We1[c] + e1 * We1[FI + c] + be1[c];
      mtile[el][c] = (__bf16)(cosv * w * hv[j]);
      e_h[(size_t)e * FI + c] = (__bf16)fmaxf(w, 0.0f);
    }
  }
  __syncthreads();

  const int wave = tid >> 5, lane = tid & 31;
  const int col = lane & 15, half = lane >> 4;
  const int n0 = wave * 16;

  // Phase B: r1 = relu(m @ Wr1a + br1a)
  {
    bf16x16 bfr[4];
#pragma unroll
    for (int kt = 0; kt < 4; ++kt) bfr[kt] = load_b_frag(Wr1a_t, FI, n0, kt * 32, lane);
    const float bias = br1a[n0 + col];
#pragma unroll
    for (int mt = 0; mt < 2; ++mt) {
      f32x8 acc = {};
#pragma unroll
      for (int kt = 0; kt < 4; ++kt)
        acc = wmma_bf(load_a_frag(&mtile[mt * 16][0] + kt * 32, FI, lane), bfr[kt], acc);
#pragma unroll
      for (int v = 0; v < 8; ++v) {
        const int row = mt * 16 + half * 8 + v;
        rtile[row][n0 + col] = (__bf16)fmaxf(acc[v] + bias, 0.0f);
      }
    }
  }
  __syncthreads();

  // Phase C: r2 = relu(r1 @ Wr1b + br1b), atomic segment-sum into s1[dst]
  {
    bf16x16 bfr[4];
#pragma unroll
    for (int kt = 0; kt < 4; ++kt) bfr[kt] = load_b_frag(Wr1b_t, FI, n0, kt * 32, lane);
    const float bias = br1b[n0 + col];
#pragma unroll
    for (int mt = 0; mt < 2; ++mt) {
      f32x8 acc = {};
#pragma unroll
      for (int kt = 0; kt < 4; ++kt)
        acc = wmma_bf(load_a_frag(&rtile[mt * 16][0] + kt * 32, FI, lane), bfr[kt], acc);
#pragma unroll
      for (int v = 0; v < 8; ++v) {
        const int row = mt * 16 + half * 8 + v;
        const int d = sdst[row];
        atomicAdd(&s1[(size_t)d * FI + n0 + col], fmaxf(acc[v] + bias, 0.0f));
      }
    }
  }
}

// ---------------- Layer 1 node kernel: h1 = relu(concat(h0, s1/cnt) @ Wl1 + bl1) ----
// 32 nodes/block, K = 256, N = 256 (8 waves x 2 n-tiles).
__global__ __launch_bounds__(256) void l1_node(
    const float* __restrict__ node_feat, const float* __restrict__ s1,
    const float* __restrict__ cnt,
    const __bf16* __restrict__ Wl1_t, const float* __restrict__ bl1,
    __bf16* __restrict__ h1)                  // [NN,256] bf16
{
  __shared__ __bf16 atile[32][2 * FI];   // 16 KB
  const int tid = threadIdx.x;
  const int nbase = blockIdx.x * 32;
  {
    const int rl = tid >> 3, part = tid & 7;   // 32 cols each
    const int node = nbase + rl;
    if (node < NN) {
      const float inv = 1.0f / fmaxf(cnt[node], 1.0f);
#pragma unroll
      for (int j = 0; j < 32; ++j) {
        const int c = part * 32 + j;
        const float x = (c < FI) ? node_feat[(size_t)node * FI + c]
                                 : s1[(size_t)node * FI + (c - FI)] * inv;
        atile[rl][c] = (__bf16)x;
      }
    } else {
#pragma unroll
      for (int j = 0; j < 32; ++j) atile[rl][part * 32 + j] = (__bf16)0.0f;
    }
  }
  __syncthreads();

  const int wave = tid >> 5, lane = tid & 31;
  const int col = lane & 15, half = lane >> 4;
#pragma unroll
  for (int nt2 = 0; nt2 < 2; ++nt2) {
    const int n0 = (wave * 2 + nt2) * 16;
    bf16x16 bfr[8];
#pragma unroll
    for (int kt = 0; kt < 8; ++kt) bfr[kt] = load_b_frag(Wl1_t, 2 * FI, n0, kt * 32, lane);
    const float bias = bl1[n0 + col];
#pragma unroll
    for (int mt = 0; mt < 2; ++mt) {
      f32x8 acc = {};
#pragma unroll
      for (int kt = 0; kt < 8; ++kt)
        acc = wmma_bf(load_a_frag(&atile[mt * 16][0] + kt * 32, 2 * FI, lane), bfr[kt], acc);
#pragma unroll
      for (int v = 0; v < 8; ++v) {
        const int node = nbase + mt * 16 + half * 8 + v;
        if (node < NN)
          h1[(size_t)node * HD + n0 + col] = (__bf16)fmaxf(acc[v] + bias, 0.0f);
      }
    }
  }
}

// ---------------- Layer 2 fused edge kernel ----------------
// 32 edges/block. w2 = e_h @ We2 + be2 (K=128,N=256); m = cos*w2*hs;
// r = relu(m@Wr2a+b); relu(r@Wr2b+b) -> atomic s2[dst].
__global__ __launch_bounds__(256) void l2_edge(
    const __bf16* __restrict__ h1,            // [NN,256]
    const __bf16* __restrict__ e_h,           // [NE,128]
    const int* __restrict__ src, const int* __restrict__ dst,
    const __bf16* __restrict__ We2_t, const float* __restrict__ be2,
    const __bf16* __restrict__ Wr2a_t, const float* __restrict__ br2a,
    const __bf16* __restrict__ Wr2b_t, const float* __restrict__ br2b,
    float* __restrict__ s2)                   // [NN,256] accum
{
  __shared__ __bf16 mtile[32][HD];   // 16 KB
  __shared__ __bf16 rtile[32][HD];   // 16 KB
  __shared__ float  scos[32];
  __shared__ int    ssrc[32], sdst2[32];

  const int tid = threadIdx.x;
  const int ebase = blockIdx.x * 32;

  // Phase A: cosine over 256-wide bf16 features (8 threads/edge, 32 cols each)
  {
    const int el = tid >> 3, part = tid & 7;
    const int e = ebase + el;
    const int s = src[e], d = dst[e];
    if (part == 0) { ssrc[el] = s; sdst2[el] = d; }
    const __bf16* hs = h1 + (size_t)s * HD + part * 32;
    const __bf16* hd = h1 + (size_t)d * HD + part * 32;
    float ss = 0.f, dd = 0.f, sd = 0.f;
#pragma unroll
    for (int q = 0; q < 4; ++q) {
      bf16x8 a = *(const bf16x8*)(hs + q * 8);
      bf16x8 b = *(const bf16x8*)(hd + q * 8);
#pragma unroll
      for (int j = 0; j < 8; ++j) {
        const float x = (float)a[j], y = (float)b[j];
        ss += x * x; dd += y * y; sd += x * y;
      }
    }
#pragma unroll
    for (int m = 1; m < 8; m <<= 1) {
      ss += __shfl_xor(ss, m, 32);
      dd += __shfl_xor(dd, m, 32);
      sd += __shfl_xor(sd, m, 32);
    }
    if (part == 0)
      scos[el] = sd / (fmaxf(sqrtf(ss), EPSF) * fmaxf(sqrtf(dd), EPSF));
  }
  __syncthreads();

  const int wave = tid >> 5, lane = tid & 31;
  const int col = lane & 15, half = lane >> 4;

  // Phase B1: w2 GEMM (A = e_h tile directly from global, row-major, L2-hot)
#pragma unroll
  for (int nt2 = 0; nt2 < 2; ++nt2) {
    const int n0 = (wave * 2 + nt2) * 16;
    bf16x16 bfr[4];
#pragma unroll
    for (int kt = 0; kt < 4; ++kt) bfr[kt] = load_b_frag(We2_t, FI, n0, kt * 32, lane);
    const float bias = be2[n0 + col];
#pragma unroll
    for (int mt = 0; mt < 2; ++mt) {
      f32x8 acc = {};
      const __bf16* abase = e_h + (size_t)(ebase + mt * 16) * FI;
#pragma unroll
      for (int kt = 0; kt < 4; ++kt)
        acc = wmma_bf(load_a_frag(abase + kt * 32, FI, lane), bfr[kt], acc);
#pragma unroll
      for (int v = 0; v < 8; ++v) {
        const int row = mt * 16 + half * 8 + v;
        const float hsv = (float)h1[(size_t)ssrc[row] * HD + n0 + col];
        mtile[row][n0 + col] = (__bf16)(scos[row] * (acc[v] + bias) * hsv);
      }
    }
  }
  __syncthreads();

  // Phase B2: r = relu(m @ Wr2a + br2a)
#pragma unroll
  for (int nt2 = 0; nt2 < 2; ++nt2) {
    const int n0 = (wave * 2 + nt2) * 16;
    bf16x16 bfr[8];
#pragma unroll
    for (int kt = 0; kt < 8; ++kt) bfr[kt] = load_b_frag(Wr2a_t, HD, n0, kt * 32, lane);
    const float bias = br2a[n0 + col];
#pragma unroll
    for (int mt = 0; mt < 2; ++mt) {
      f32x8 acc = {};
#pragma unroll
      for (int kt = 0; kt < 8; ++kt)
        acc = wmma_bf(load_a_frag(&mtile[mt * 16][0] + kt * 32, HD, lane), bfr[kt], acc);
#pragma unroll
      for (int v = 0; v < 8; ++v) {
        const int row = mt * 16 + half * 8 + v;
        rtile[row][n0 + col] = (__bf16)fmaxf(acc[v] + bias, 0.0f);
      }
    }
  }
  __syncthreads();

  // Phase B3: relu(r @ Wr2b + br2b) -> atomic segment-sum
#pragma unroll
  for (int nt2 = 0; nt2 < 2; ++nt2) {
    const int n0 = (wave * 2 + nt2) * 16;
    bf16x16 bfr[8];
#pragma unroll
    for (int kt = 0; kt < 8; ++kt) bfr[kt] = load_b_frag(Wr2b_t, HD, n0, kt * 32, lane);
    const float bias = br2b[n0 + col];
#pragma unroll
    for (int mt = 0; mt < 2; ++mt) {
      f32x8 acc = {};
#pragma unroll
      for (int kt = 0; kt < 8; ++kt)
        acc = wmma_bf(load_a_frag(&rtile[mt * 16][0] + kt * 32, HD, lane), bfr[kt], acc);
#pragma unroll
      for (int v = 0; v < 8; ++v) {
        const int row = mt * 16 + half * 8 + v;
        const int d = sdst2[row];
        atomicAdd(&s2[(size_t)d * HD + n0 + col], fmaxf(acc[v] + bias, 0.0f));
      }
    }
  }
}

// ---------------- Layer 2 node kernel: out = concat(h1, s2/cnt) @ Wl2 + bl2 ----
// 32 nodes/block, K = 512, N = 32. Waves 0..3 do (mt, nt) tiles.
__global__ __launch_bounds__(256) void l2_node(
    const __bf16* __restrict__ h1, const float* __restrict__ s2,
    const float* __restrict__ cnt,
    const __bf16* __restrict__ Wl2_t, const float* __restrict__ bl2,
    float* __restrict__ out)                  // [NN,32]
{
  __shared__ __bf16 atile[32][2 * HD];   // 32 KB
  const int tid = threadIdx.x;
  const int nbase = blockIdx.x * 32;
  {
    const int rl = tid >> 3, part = tid & 7;   // 64 cols each
    const int node = nbase + rl;
    if (node < NN) {
      const float inv = 1.0f / fmaxf(cnt[node], 1.0f);
#pragma unroll
      for (int j = 0; j < 64; ++j) {
        const int c = part * 64 + j;
        const float x = (c < HD) ? (float)h1[(size_t)node * HD + c]
                                 : s2[(size_t)node * HD + (c - HD)] * inv;
        atile[rl][c] = (__bf16)x;
      }
    } else {
#pragma unroll
      for (int j = 0; j < 64; ++j) atile[rl][part * 64 + j] = (__bf16)0.0f;
    }
  }
  __syncthreads();

  const int wave = tid >> 5, lane = tid & 31;
  if (wave < 4) {
    const int col = lane & 15, half = lane >> 4;
    const int mt = wave >> 1, n0 = (wave & 1) * 16;
    f32x8 acc = {};
#pragma unroll
    for (int kt = 0; kt < 16; ++kt) {
      bf16x16 a = load_a_frag(&atile[mt * 16][0] + kt * 32, 2 * HD, lane);
      bf16x16 b = load_b_frag(Wl2_t, 2 * HD, n0, kt * 32, lane);
      acc = wmma_bf(a, b, acc);
    }
    const float bias = bl2[n0 + col];
#pragma unroll
    for (int v = 0; v < 8; ++v) {
      const int node = nbase + mt * 16 + half * 8 + v;
      if (node < NN) out[(size_t)node * CD + n0 + col] = acc[v] + bias;
    }
  }
}

// ---------------- Host launch ----------------

extern "C" void kernel_launch(void* const* d_in, const int* in_sizes, int n_in,
                              void* d_out, int out_size, void* d_ws, size_t ws_size,
                              hipStream_t stream) {
  (void)in_sizes; (void)n_in; (void)out_size; (void)ws_size;
  const float* node_feat = (const float*)d_in[0];
  const float* edge_feat = (const float*)d_in[1];
  const int*   src       = (const int*)d_in[2];
  const int*   dst       = (const int*)d_in[3];
  const float* We1  = (const float*)d_in[4];  const float* be1  = (const float*)d_in[5];
  const float* Wr1a = (const float*)d_in[6];  const float* br1a = (const float*)d_in[7];
  const float* Wr1b = (const float*)d_in[8];  const float* br1b = (const float*)d_in[9];
  const float* Wl1  = (const float*)d_in[10]; const float* bl1  = (const float*)d_in[11];
  const float* We2  = (const float*)d_in[12]; const float* be2  = (const float*)d_in[13];
  const float* Wr2a = (const float*)d_in[14]; const float* br2a = (const float*)d_in[15];
  const float* Wr2b = (const float*)d_in[16]; const float* br2b = (const float*)d_in[17];
  const float* Wl2  = (const float*)d_in[18]; const float* bl2  = (const float*)d_in[19];

  char* ws = (char*)d_ws;
  size_t off = 0;
  auto carve = [&](size_t bytes) -> char* {
    char* p = ws + off;
    off = (off + bytes + 255) & ~(size_t)255;
    return p;
  };
  // zeroed region first (s1 | s2 | cnt contiguous)
  float* s1  = (float*)carve((size_t)NN * FI * 4);
  float* s2  = (float*)carve((size_t)NN * HD * 4);
  float* cnt = (float*)carve((size_t)NN * 4);
  const long zero_n = (long)(off / 4);
  __bf16* e_h   = (__bf16*)carve((size_t)NE * FI * 2);
  __bf16* h1    = (__bf16*)carve((size_t)NN * HD * 2);
  __bf16* Wr1a_t = (__bf16*)carve((size_t)FI * FI * 2);
  __bf16* Wr1b_t = (__bf16*)carve((size_t)FI * FI * 2);
  __bf16* Wl1_t  = (__bf16*)carve((size_t)2 * FI * HD * 2);
  __bf16* We2_t  = (__bf16*)carve((size_t)FI * HD * 2);
  __bf16* Wr2a_t = (__bf16*)carve((size_t)HD * HD * 2);
  __bf16* Wr2b_t = (__bf16*)carve((size_t)HD * HD * 2);
  __bf16* Wl2_t  = (__bf16*)carve((size_t)2 * HD * CD * 2);

  // 1) zero accumulators + degree counts
  zero_f32<<<4096, 256, 0, stream>>>((float*)d_ws, zero_n);

  // 2) weight conversion to N-major bf16
  auto tconv = [&](const float* w, __bf16* wt, int K, int N) {
    w_to_bf16_t<<<(K * N + 255) / 256, 256, 0, stream>>>(w, wt, K, N);
  };
  tconv(Wr1a, Wr1a_t, FI, FI);
  tconv(Wr1b, Wr1b_t, FI, FI);
  tconv(Wl1,  Wl1_t,  2 * FI, HD);
  tconv(We2,  We2_t,  FI, HD);
  tconv(Wr2a, Wr2a_t, HD, HD);
  tconv(Wr2b, Wr2b_t, HD, HD);
  tconv(Wl2,  Wl2_t,  2 * HD, CD);

  // 3) in-degree counts (shared by both layers)
  count_edges<<<(NE + 255) / 256, 256, 0, stream>>>(dst, cnt);

  // 4) layer 1 fused edge pass
  l1_edge<<<NE / 32, 256, 0, stream>>>(node_feat, edge_feat, src, dst,
                                       We1, be1, Wr1a_t, br1a, Wr1b_t, br1b,
                                       s1, e_h);
  // 5) layer 1 node update
  l1_node<<<(NN + 31) / 32, 256, 0, stream>>>(node_feat, s1, cnt, Wl1_t, bl1, h1);

  // 6) layer 2 fused edge pass
  l2_edge<<<NE / 32, 256, 0, stream>>>(h1, e_h, src, dst,
                                       We2_t, be2, Wr2a_t, br2a, Wr2b_t, br2b, s2);
  // 7) layer 2 node update -> logits
  l2_node<<<(NN + 31) / 32, 256, 0, stream>>>(h1, s2, cnt, Wl2_t, bl2,
                                              (float*)d_out);
}
